// GeneMamba_70325794504910
// MI455X (gfx1250) — matrix operand order repaired and checked by reading, not compiled, not run on hardware
//
#include <hip/hip_runtime.h>
#include <hip/hip_bf16.h>
#include <cstdint>

// ---------------------------------------------------------------------------
// Mamba-2 forward for MI455X (gfx1250, wave32, WMMA bf16 -> f32 accumulate,
// async global->LDS staging for the GEMMs)
// ---------------------------------------------------------------------------

typedef __attribute__((ext_vector_type(16))) __bf16 v16bf;
typedef __attribute__((ext_vector_type(8)))  float  v8f;

#define D_MODEL   768
#define D_STATE   64
#define D_INNER   1536
#define NHEADS    24
#define HEADDIM   64
#define CHUNK     64
#define NCHUNK    64          // 4096 / 64
#define SEQ       4096
#define BSZ       8
#define D_IN_PROJ 3224        // 2*1536 + 2*64 + 24
#define CONV_DIM  1664        // 1536 + 128
#define TOKENS    32768L      // 8*4096

#ifndef __has_builtin
#define __has_builtin(x) 0
#endif

#if __has_builtin(__builtin_amdgcn_global_load_async_to_lds_b128)
#define USE_ASYNC_LDS 1
#else
#define USE_ASYNC_LDS 0
#endif

#if USE_ASYNC_LDS
typedef int v4i_async __attribute__((vector_size(16)));
typedef __attribute__((address_space(1))) v4i_async* gv4i_ptr;
typedef __attribute__((address_space(3))) v4i_async* lv4i_ptr;

__device__ inline void async_cp16(void* lds, const void* g) {
  // global_load_async_to_lds_b128: 16B per lane, tracked by ASYNCcnt.
  // Flat pointers to LDS carry the LDS byte offset in addr[31:0].
  __builtin_amdgcn_global_load_async_to_lds_b128(
      (gv4i_ptr)(uintptr_t)g,
      (lv4i_ptr)(unsigned)(uintptr_t)lds,
      0, 0);
}
__device__ inline void wait_async4() {
#if __has_builtin(__builtin_amdgcn_s_wait_asynccnt)
  __builtin_amdgcn_s_wait_asynccnt(4);
#else
  asm volatile("s_wait_asynccnt 0x4" ::: "memory");
#endif
}
__device__ inline void wait_async0() {
#if __has_builtin(__builtin_amdgcn_s_wait_asynccnt)
  __builtin_amdgcn_s_wait_asynccnt(0);
#else
  asm volatile("s_wait_asynccnt 0x0" ::: "memory");
#endif
}
#else
__device__ inline void wait_async4() {}
__device__ inline void wait_async0() {}
#endif

__device__ inline v8f v8f_zero() {
  v8f z;
#pragma unroll
  for (int i = 0; i < 8; ++i) z[i] = 0.f;
  return z;
}

// A-matrix 16x32 bf16 fragment (CDNA5 layout): lanes 0-15 rows M=0..15,
// lane half h: elems 0..7 -> K = h*8 + i ; elems 8..15 -> K = 16 + h*8 + i.
__device__ inline v16bf frag_a(const __bf16* row, int h) {
  v16bf a;
#pragma unroll
  for (int i = 0; i < 8; ++i) a[i] = row[h * 8 + i];
#pragma unroll
  for (int i = 0; i < 8; ++i) a[8 + i] = row[16 + h * 8 + i];
  return a;
}

// B-matrix 32x16 bf16 fragment: col n = lane&15; lane half h:
// elems 0..7 -> K = h*8 + i ; elems 8..15 -> K = 16 + h*8 + i.
__device__ inline v16bf frag_b(const __bf16* row, int h) {
  v16bf b;
#pragma unroll
  for (int i = 0; i < 8; ++i) b[i] = row[h * 8 + i];
#pragma unroll
  for (int i = 0; i < 8; ++i) b[8 + i] = row[16 + h * 8 + i];
  return b;
}

__device__ inline v8f wmma_bf16(v16bf a, v16bf b, v8f c) {
  return __builtin_amdgcn_wmma_f32_16x16x32_bf16(
      false, a, false, b, (short)0, c, false, false);
}

__device__ inline float sigmoidf_(float x) { return 1.f / (1.f + __expf(-x)); }

// ---------------------------------------------------------------------------
// fp32 -> bf16 bulk convert (n multiple of 4)
// ---------------------------------------------------------------------------
__global__ __launch_bounds__(256) void k_cvt_bf16(const float* __restrict__ src,
                                                  __bf16* __restrict__ dst,
                                                  long n) {
  long i = ((long)blockIdx.x * 256 + threadIdx.x) * 4;
  if (i >= n) return;
  float4 v = *(const float4*)(src + i);
  __bf16 o[4] = {(__bf16)v.x, (__bf16)v.y, (__bf16)v.z, (__bf16)v.w};
  *(uint2*)(dst + i) = *(const uint2*)o;
}

// ---------------------------------------------------------------------------
// C[M,N] = A[M,K] * B[N,K]^T ; A,B bf16 row-major, C fp32.
// 256 threads = 8 waves; block tile 128x128; wave tile 32x64 (8 WMMA accs).
// Double-buffered LDS, filled with async global->LDS b128 copies.
// K % 32 == 0, M % 128 == 0, N edge-guarded (B rows clamped, stores masked).
// ---------------------------------------------------------------------------
#define LDA_S 40   // padded LDS row stride (bf16 elems) -> conflict-free

__global__ __launch_bounds__(256) void k_gemm_bf16(
    const __bf16* __restrict__ A, const __bf16* __restrict__ B,
    float* __restrict__ C, int M, int N, int K) {
  __shared__ __align__(16) __bf16 As[2][128 * LDA_S];
  __shared__ __align__(16) __bf16 Bs[2][128 * LDA_S];
  const int t = threadIdx.x;
  const int lane = t & 31, h = lane >> 4, l15 = lane & 15, w = t >> 5;
  const int wm = (w & 3) * 32;       // wave M offset in block tile
  const int wn = (w >> 2) * 64;      // wave N offset in block tile
  const int bm = blockIdx.x * 128;
  const int bn = blockIdx.y * 128;

  // Per-thread staging slots: tile = 128 rows x 32 cols bf16 = 512 x 16B units,
  // 4 units per row; 2 units per thread for A and for B.
  const __bf16* Aga[2];
  const __bf16* Bga[2];
  unsigned ldsOff[2];
#pragma unroll
  for (int i = 0; i < 2; ++i) {
    int u2 = t + i * 256;
    int row = u2 >> 2;
    int uc = (u2 & 3) * 8;          // bf16 element offset within row
    ldsOff[i] = row * LDA_S + uc;
    Aga[i] = A + (size_t)(bm + row) * K + uc;
    int brow = bn + row;
    if (brow >= N) brow = 0;        // clamp: junk lands in unused cols only
    Bga[i] = B + (size_t)brow * K + uc;
  }

  v8f acc[2][4];
#pragma unroll
  for (int i = 0; i < 2; ++i)
#pragma unroll
    for (int j = 0; j < 4; ++j) acc[i][j] = v8f_zero();

  auto issue = [&](int k0, int buf) {
#pragma unroll
    for (int i = 0; i < 2; ++i) {
#if USE_ASYNC_LDS
      async_cp16(&As[buf][ldsOff[i]], Aga[i] + k0);
      async_cp16(&Bs[buf][ldsOff[i]], Bga[i] + k0);
#else
      *(uint4*)&As[buf][ldsOff[i]] = *(const uint4*)(Aga[i] + k0);
      *(uint4*)&Bs[buf][ldsOff[i]] = *(const uint4*)(Bga[i] + k0);
#endif
    }
  };

  const int nk = K >> 5;
  issue(0, 0);
  for (int kk = 0; kk < nk; ++kk) {
    const int buf = kk & 1;
    if (kk + 1 < nk) {
      issue((kk + 1) << 5, buf ^ 1);   // prefetch next stage
      wait_async4();                   // retire current stage's 4 ops
    } else {
      wait_async0();
    }
    __syncthreads();

    v16bf a0 = frag_a(&As[buf][(wm + l15) * LDA_S], h);
    v16bf a1 = frag_a(&As[buf][(wm + 16 + l15) * LDA_S], h);
#pragma unroll
    for (int tj = 0; tj < 4; ++tj) {
      v16bf bb = frag_b(&Bs[buf][(wn + tj * 16 + l15) * LDA_S], h);
      acc[0][tj] = wmma_bf16(a0, bb, acc[0][tj]);
      acc[1][tj] = wmma_bf16(a1, bb, acc[1][tj]);
    }
    __syncthreads();
  }

#pragma unroll
  for (int ti = 0; ti < 2; ++ti)
#pragma unroll
    for (int tj = 0; tj < 4; ++tj) {
      int col = bn + wn + tj * 16 + l15;
      if (col < N) {
#pragma unroll
        for (int r = 0; r < 8; ++r) {
          int m = bm + wm + ti * 16 + h * 8 + r;
          C[(size_t)m * N + col] = acc[ti][tj][r];
        }
      }
    }
}

// ---------------------------------------------------------------------------
// causal depthwise conv(4) + bias + SiLU over xBC channels
// ---------------------------------------------------------------------------
__global__ __launch_bounds__(256) void k_conv_silu(
    const float* __restrict__ zxbcdt, const float* __restrict__ conv_w,
    const float* __restrict__ conv_b, float* __restrict__ xBCc, long total) {
  long idx = (long)blockIdx.x * 256 + threadIdx.x;
  if (idx >= total) return;
  int ch = (int)(idx % CONV_DIM);
  long tok = idx / CONV_DIM;
  int l = (int)(tok % SEQ);
  float acc = conv_b[ch];
#pragma unroll
  for (int j = 0; j < 4; ++j) {
    int ll = l - 3 + j;
    if (ll >= 0) {
      float v = zxbcdt[(tok - 3 + j) * D_IN_PROJ + D_INNER + ch];
      acc += conv_w[ch * 4 + j] * v;
    }
  }
  xBCc[idx] = acc * sigmoidf_(acc);
}

// ---------------------------------------------------------------------------
// dt = softplus(raw + dt_bias); adt = -exp(A_log)*dt
// ---------------------------------------------------------------------------
__global__ __launch_bounds__(256) void k_dt(
    const float* __restrict__ zxbcdt, const float* __restrict__ dt_bias,
    const float* __restrict__ A_log, float* __restrict__ dt,
    float* __restrict__ adt, long total) {
  long i = (long)blockIdx.x * 256 + threadIdx.x;
  if (i >= total) return;
  int hh = (int)(i % NHEADS);
  long tok = i / NHEADS;
  float x = zxbcdt[tok * D_IN_PROJ + D_INNER + CONV_DIM + hh] + dt_bias[hh];
  float sp = (x > 20.f) ? x : log1pf(__expf(x));
  dt[i] = sp;
  adt[i] = -__expf(A_log[hh]) * sp;
}

// ---------------------------------------------------------------------------
// per (b,c,h) chunk: Y_diag (masked CB^T @ x) and local states (WMMA x3)
// ---------------------------------------------------------------------------
#define LDC_S 72   // padded LDS stride for 64 cols

__global__ __launch_bounds__(128) void k_chunk(
    const float* __restrict__ xBCc, const float* __restrict__ dtb,
    const float* __restrict__ adtb, float* __restrict__ Ybuf,
    float* __restrict__ Sbuf, float* __restrict__ Alast) {
  const int c = blockIdx.x, hh = blockIdx.y, b = blockIdx.z;
  __shared__ __bf16 sC[64 * LDC_S];    // C[l][n]
  __shared__ __bf16 sB[64 * LDC_S];    // B[l][n]
  __shared__ __bf16 sBT[64 * LDC_S];   // B^T[n][l]
  __shared__ __bf16 sXT[64 * LDC_S];   // (x*dt)^T[p][l]
  __shared__ __bf16 sXDT[64 * LDC_S];  // (x*dt*decay)^T[p][l]
  __shared__ __bf16 sG[64 * LDC_S];    // masked decay C B^T [l][lj]
  __shared__ float sdt[64], scs[64], sdec[64];

  const int t = threadIdx.x, lane = t & 31, h = lane >> 4, l15 = lane & 15,
            w = t >> 5;
  const long tok0 = (long)b * SEQ + (long)c * CHUNK;

  if (t < 64) {
    sdt[t] = dtb[(tok0 + t) * NHEADS + hh];
    sdec[t] = adtb[(tok0 + t) * NHEADS + hh];   // stage adt in sdec
  }
  __syncthreads();
  if (t < 64) {
    float cs = 0.f;
    for (int k2 = 0; k2 <= t; ++k2) cs += sdec[k2];
    scs[t] = cs;
  }
  __syncthreads();
  float cstot = scs[63];
  if (t == 0) Alast[((long)b * NHEADS + hh) * NCHUNK + c] = cstot;
  if (t < 64) sdec[t] = __expf(cstot - scs[t]);  // decay_states[l]
  __syncthreads();

  for (int e = t; e < 4096; e += 128) {
    int l = e >> 6, n = e & 63;
    float vb = xBCc[(tok0 + l) * (long)CONV_DIM + D_INNER + n];
    float vc = xBCc[(tok0 + l) * (long)CONV_DIM + D_INNER + D_STATE + n];
    sB[l * LDC_S + n] = (__bf16)vb;
    sBT[n * LDC_S + l] = (__bf16)vb;
    sC[l * LDC_S + n] = (__bf16)vc;
  }
  for (int e = t; e < 4096; e += 128) {
    int l = e >> 6, p = e & 63;
    float xv = xBCc[(tok0 + l) * (long)CONV_DIM + hh * HEADDIM + p] * sdt[l];
    sXT[p * LDC_S + l] = (__bf16)xv;
    sXDT[p * LDC_S + l] = (__bf16)(xv * sdec[l]);
  }
  __syncthreads();

  // --- G = C @ B^T, then mask with exp(cs_i - cs_j) on lower triangle ---
  {
    v8f g[4];
#pragma unroll
    for (int i = 0; i < 4; ++i) g[i] = v8f_zero();
    const __bf16* arow = &sC[(w * 16 + l15) * LDC_S];
#pragma unroll
    for (int k0 = 0; k0 < 64; k0 += 32) {
      v16bf a = frag_a(arow + k0, h);
#pragma unroll
      for (int tj = 0; tj < 4; ++tj) {
        v16bf bb = frag_b(&sB[(tj * 16 + l15) * LDC_S + k0], h);
        g[tj] = wmma_bf16(a, bb, g[tj]);
      }
    }
#pragma unroll
    for (int tj = 0; tj < 4; ++tj)
#pragma unroll
      for (int r = 0; r < 8; ++r) {
        int li = w * 16 + h * 8 + r;
        int lj = tj * 16 + l15;
        float gv = g[tj][r];
        gv = (lj <= li) ? gv * __expf(scs[li] - scs[lj]) : 0.f;
        sG[li * LDC_S + lj] = (__bf16)gv;
      }
  }
  __syncthreads();

  // --- Y_diag = G_masked @ xd ; S = (xd*decay)^T @ B ---
  v8f y[4], s[4];
#pragma unroll
  for (int i = 0; i < 4; ++i) { y[i] = v8f_zero(); s[i] = v8f_zero(); }
  const __bf16* grow = &sG[(w * 16 + l15) * LDC_S];
  const __bf16* xrow = &sXDT[(w * 16 + l15) * LDC_S];
#pragma unroll
  for (int k0 = 0; k0 < 64; k0 += 32) {
    v16bf ag = frag_a(grow + k0, h);
    v16bf ax = frag_a(xrow + k0, h);
#pragma unroll
    for (int tj = 0; tj < 4; ++tj) {
      v16bf bx = frag_b(&sXT[(tj * 16 + l15) * LDC_S + k0], h);
      y[tj] = wmma_bf16(ag, bx, y[tj]);
      v16bf bbt = frag_b(&sBT[(tj * 16 + l15) * LDC_S + k0], h);
      s[tj] = wmma_bf16(ax, bbt, s[tj]);
    }
  }

#pragma unroll
  for (int tj = 0; tj < 4; ++tj)
#pragma unroll
    for (int r = 0; r < 8; ++r) {
      int row = w * 16 + h * 8 + r;     // l for Y, p for S
      int col = tj * 16 + l15;          // p for Y, n for S
      Ybuf[((tok0 + row) * NHEADS + hh) * HEADDIM + col] = y[tj][r];
      Sbuf[((((long)b * NHEADS + hh) * NCHUNK + c) * HEADDIM + row) * D_STATE +
           col] = s[tj][r];
    }
}

// ---------------------------------------------------------------------------
// sequential inter-chunk state recurrence per (b,h); in place on Sbuf
// ---------------------------------------------------------------------------
__global__ __launch_bounds__(256) void k_scan(float* __restrict__ Sbuf,
                                              const float* __restrict__ Alast) {
  const int bh = blockIdx.x;
  const int t = threadIdx.x;
  float s[16];
#pragma unroll
  for (int i = 0; i < 16; ++i) s[i] = 0.f;
  float* base = Sbuf + (size_t)bh * NCHUNK * (HEADDIM * D_STATE);
  const float* al = Alast + (size_t)bh * NCHUNK;
  for (int c = 0; c < NCHUNK; ++c) {
    float wv = __expf(al[c]);
    float* p = base + (size_t)c * (HEADDIM * D_STATE) + t * 16;
#pragma unroll
    for (int i = 0; i < 16; ++i) {
      float loc = p[i];
      p[i] = s[i];
      s[i] = wv * s[i] + loc;
    }
  }
}

// ---------------------------------------------------------------------------
// Y += exp(cs_l) * (C @ states^T) + D*x   (WMMA 64x64x64)
// ---------------------------------------------------------------------------
__global__ __launch_bounds__(128) void k_yoff(
    const float* __restrict__ xBCc, const float* __restrict__ adtb,
    const float* __restrict__ Sbuf, const float* __restrict__ D_param,
    float* __restrict__ Ybuf) {
  const int c = blockIdx.x, hh = blockIdx.y, b = blockIdx.z;
  __shared__ __bf16 sC[64 * LDC_S];   // C[l][n]
  __shared__ __bf16 sS[64 * LDC_S];   // states_in[p][n]
  __shared__ float scs[64], stmp[64];
  const int t = threadIdx.x, lane = t & 31, h = lane >> 4, l15 = lane & 15,
            w = t >> 5;
  const long tok0 = (long)b * SEQ + (long)c * CHUNK;

  if (t < 64) stmp[t] = adtb[(tok0 + t) * NHEADS + hh];
  __syncthreads();
  if (t < 64) {
    float cs = 0.f;
    for (int k2 = 0; k2 <= t; ++k2) cs += stmp[k2];
    scs[t] = cs;
  }
  const float* sb =
      Sbuf + (((size_t)b * NHEADS + hh) * NCHUNK + c) * (HEADDIM * D_STATE);
  for (int e = t; e < 4096; e += 128) {
    int l = e >> 6, n = e & 63;
    sC[l * LDC_S + n] =
        (__bf16)xBCc[(tok0 + l) * (long)CONV_DIM + D_INNER + D_STATE + n];
    sS[l * LDC_S + n] = (__bf16)sb[e];  // here l plays the role of p
  }
  __syncthreads();

  v8f y[4];
#pragma unroll
  for (int i = 0; i < 4; ++i) y[i] = v8f_zero();
  const __bf16* arow = &sC[(w * 16 + l15) * LDC_S];
#pragma unroll
  for (int k0 = 0; k0 < 64; k0 += 32) {
    v16bf a = frag_a(arow + k0, h);
#pragma unroll
    for (int tj = 0; tj < 4; ++tj) {
      v16bf bb = frag_b(&sS[(tj * 16 + l15) * LDC_S + k0], h);
      y[tj] = wmma_bf16(a, bb, y[tj]);
    }
  }

  const float Dp = D_param[hh];
#pragma unroll
  for (int tj = 0; tj < 4; ++tj)
#pragma unroll
    for (int r = 0; r < 8; ++r) {
      int l = w * 16 + h * 8 + r;
      int p = tj * 16 + l15;
      size_t oi = ((tok0 + l) * NHEADS + hh) * HEADDIM + p;
      float xraw = xBCc[(tok0 + l) * (long)CONV_DIM + hh * HEADDIM + p];
      Ybuf[oi] = Ybuf[oi] + __expf(scs[l]) * y[tj][r] + Dp * xraw;
    }
}

// ---------------------------------------------------------------------------
// SiLU(z) gating + RMSNorm over D_INNER; writes bf16 for the out-proj GEMM.
// ---------------------------------------------------------------------------
__global__ __launch_bounds__(256) void k_gatenorm(
    const float* __restrict__ zxbcdt, const float* __restrict__ norm_w,
    const float* __restrict__ Ybuf, __bf16* __restrict__ ybf) {
  const long tok = blockIdx.x;
  const int t = threadIdx.x;
  __shared__ float red[256];
  float vals[6];
  float ss = 0.f;
#pragma unroll
  for (int i = 0; i < 6; ++i) {
    int idx = t + i * 256;
    float z = zxbcdt[tok * D_IN_PROJ + idx];
    float y = Ybuf[tok * D_INNER + idx];
    float g = y * (z * sigmoidf_(z));
    vals[i] = g;
    ss += g * g;
  }
  red[t] = ss;
  __syncthreads();
  for (int s2 = 128; s2 > 0; s2 >>= 1) {
    if (t < s2) red[t] += red[t + s2];
    __syncthreads();
  }
  float inv = rsqrtf(red[0] / (float)D_INNER + 1e-5f);
#pragma unroll
  for (int i = 0; i < 6; ++i) {
    int idx = t + i * 256;
    ybf[tok * D_INNER + idx] = (__bf16)(vals[i] * inv * norm_w[idx]);
  }
}

// ---------------------------------------------------------------------------
extern "C" void kernel_launch(void* const* d_in, const int* in_sizes, int n_in,
                              void* d_out, int out_size, void* d_ws,
                              size_t ws_size, hipStream_t stream) {
  const float* u       = (const float*)d_in[0];
  const float* W_in    = (const float*)d_in[1];
  const float* conv_w  = (const float*)d_in[2];
  const float* conv_b  = (const float*)d_in[3];
  const float* dt_bias = (const float*)d_in[4];
  const float* A_log   = (const float*)d_in[5];
  const float* D_par   = (const float*)d_in[6];
  const float* norm_w  = (const float*)d_in[7];
  const float* W_out   = (const float*)d_in[8];
  float* out = (float*)d_out;

  char* ws = (char*)d_ws;
  size_t off = 0;
  auto alloc = [&](size_t bytes) {
    void* p = ws + off;
    off += (bytes + 255) & ~(size_t)255;
    return p;
  };
  float*  zxbcdt = (float*)alloc(TOKENS * D_IN_PROJ * 4);
  float*  xBCc   = (float*)alloc(TOKENS * CONV_DIM * 4);
  float*  dtb    = (float*)alloc(TOKENS * NHEADS * 4);
  float*  adtb   = (float*)alloc(TOKENS * NHEADS * 4);
  float*  Ybuf   = (float*)alloc(TOKENS * D_INNER * 4);
  float*  Sbuf   = (float*)alloc((size_t)BSZ * NHEADS * NCHUNK * HEADDIM * D_STATE * 4);
  float*  Alast  = (float*)alloc((size_t)BSZ * NHEADS * NCHUNK * 4);
  __bf16* ubf    = (__bf16*)alloc(TOKENS * D_MODEL * 2);
  __bf16* wbf    = (__bf16*)alloc((size_t)D_IN_PROJ * D_MODEL * 2);
  __bf16* wobf   = (__bf16*)alloc((size_t)D_MODEL * D_INNER * 2);
  __bf16* ybf    = (__bf16*)alloc(TOKENS * D_INNER * 2);
  (void)ws_size; (void)in_sizes; (void)n_in; (void)out_size;

  auto cvt = [&](const float* s, __bf16* d, long n) {
    k_cvt_bf16<<<(unsigned)((n / 4 + 255) / 256), 256, 0, stream>>>(s, d, n);
  };
  // 0) pre-convert GEMM operands to bf16
  cvt(u, ubf, TOKENS * D_MODEL);
  cvt(W_in, wbf, (long)D_IN_PROJ * D_MODEL);
  cvt(W_out, wobf, (long)D_MODEL * D_INNER);

  // 1) in_proj: zxbcdt = u @ W_in^T
  {
    dim3 grid(TOKENS / 128, (D_IN_PROJ + 127) / 128);
    k_gemm_bf16<<<grid, 256, 0, stream>>>(ubf, wbf, zxbcdt, (int)TOKENS,
                                          D_IN_PROJ, D_MODEL);
  }
  // 2) conv + silu
  {
    long total = TOKENS * CONV_DIM;
    k_conv_silu<<<(unsigned)((total + 255) / 256), 256, 0, stream>>>(
        zxbcdt, conv_w, conv_b, xBCc, total);
  }
  // 3) dt / A*dt
  {
    long total = TOKENS * NHEADS;
    k_dt<<<(unsigned)((total + 255) / 256), 256, 0, stream>>>(
        zxbcdt, dt_bias, A_log, dtb, adtb, total);
  }
  // 4) per-chunk WMMA: Y_diag + local states
  {
    dim3 grid(NCHUNK, NHEADS, BSZ);
    k_chunk<<<grid, 128, 0, stream>>>(xBCc, dtb, adtb, Ybuf, Sbuf, Alast);
  }
  // 5) inter-chunk scan
  k_scan<<<BSZ * NHEADS, 256, 0, stream>>>(Sbuf, Alast);
  // 6) Y_off + D*x
  {
    dim3 grid(NCHUNK, NHEADS, BSZ);
    k_yoff<<<grid, 128, 0, stream>>>(xBCc, adtb, Sbuf, D_par, Ybuf);
  }
  // 7) gating + RMSNorm -> bf16
  k_gatenorm<<<(unsigned)TOKENS, 256, 0, stream>>>(zxbcdt, norm_w, Ybuf, ybf);
  // 8) out_proj: out = ynorm @ W_out^T
  {
    dim3 grid(TOKENS / 128, D_MODEL / 128);
    k_gemm_bf16<<<grid, 256, 0, stream>>>(ybf, wobf, out, (int)TOKENS,
                                          D_MODEL, D_INNER);
  }
}
